// MultiLocalCosineLinear_87823491269051
// MI455X (gfx1250) — matrix operand order repaired and checked by reading, not compiled, not run on hardware
//
#include <hip/hip_runtime.h>
#include <hip/hip_bf16.h>

typedef __attribute__((ext_vector_type(16))) _Float16 v16h;
typedef __attribute__((ext_vector_type(8)))  _Float16 v8h;
typedef __attribute__((ext_vector_type(2)))  __fp16   v2hf;   // cvt_pkrtz result type
typedef __attribute__((ext_vector_type(8)))  float    v8f;
typedef __attribute__((ext_vector_type(4)))  float    v4f;    // native vec4 for NT loads

#define EPS   1e-12f
#define DIM   768
#define NCLS  100
#define ROWS  16     // x rows per block tile
#define CH    256    // K-chunk staged in LDS (DIM/CH chunks)
#define NCH   (DIM / CH)

__device__ __forceinline__ v8f wmma16(v16h a, v16h b, v8f c) {
  // D = A(16xK f16) * B(Kx16 f16) + C(f32), K=32
  return __builtin_amdgcn_wmma_f32_16x16x32_f16(false, a, false, b, (short)0, c,
                                                false, false);
}

__device__ __forceinline__ unsigned asu32(v2hf h) {
  union { v2hf h2; unsigned u; } u_;
  u_.h2 = h;
  return u_.u;
}

// hi/lo split of 4 floats using v_cvt_pk_f16_f32 (2 elems/instruction).
// RTZ on hi is fine: lo = v - hi compensates (~2^-21 relative accuracy).
__device__ __forceinline__ void split4(v4f v, uint2& hi, uint2& lo) {
  v2hf h01 = __builtin_amdgcn_cvt_pkrtz(v[0], v[1]);
  v2hf h23 = __builtin_amdgcn_cvt_pkrtz(v[2], v[3]);
  v2hf l01 = __builtin_amdgcn_cvt_pkrtz(v[0] - (float)h01[0], v[1] - (float)h01[1]);
  v2hf l23 = __builtin_amdgcn_cvt_pkrtz(v[2] - (float)h23[0], v[3] - (float)h23[1]);
  hi = make_uint2(asu32(h01), asu32(h23));
  lo = make_uint2(asu32(l01), asu32(l23));
}

// ---------------------------------------------------------------------------
// Kernel 1: scale[p*2+j] = sigma[p] / max(||W[p,j,:]||, EPS)   (9900 rows)
// ---------------------------------------------------------------------------
__global__ void wscale_kernel(const float* __restrict__ W,
                              const float* __restrict__ sigma,
                              float* __restrict__ scale, int nrows) {
  int wid  = blockIdx.x * (blockDim.x >> 5) + (threadIdx.x >> 5);
  int lane = threadIdx.x & 31;
  if (wid >= nrows) return;
  const float* row = W + (size_t)wid * DIM;
  float s = 0.f;
  for (int k = lane * 4; k < DIM; k += 32 * 4) {
    v4f v = *(const v4f*)&row[k];
    s += v[0] * v[0] + v[1] * v[1] + v[2] * v[2] + v[3] * v[3];
  }
  #pragma unroll
  for (int m = 16; m >= 1; m >>= 1) s += __shfl_xor(s, m, 32);
  if (lane == 0) scale[wid] = sigma[wid >> 1] / fmaxf(sqrtf(s), EPS);
}

// ---------------------------------------------------------------------------
// Kernel 2: main. 256 threads handle 16 rows. Per tile:
//   - top-2 of first_out via 16-lane shuffle reduction -> pair_idx
//   - stage x (NT stream) and gathered W (L2-resident) as hi/lo f16 in LDS:
//     all 12 b128 loads issued before any conversion (full per-wave MLP)
//   - per wave: WMMA f32_16x16x32_f16; diag(C0)=x.W0, diag(C1)=x.W1,
//     diag(C2)=||x||^2 (hi/lo split: hh + hl + lh)
// ---------------------------------------------------------------------------
__global__ __launch_bounds__(256)
void mlcl_kernel(const float* __restrict__ x,
                 const float* __restrict__ fo,
                 const float* __restrict__ W,
                 const float* __restrict__ scale,
                 float* __restrict__ out, int B) {
  __shared__ alignas(32) _Float16 s_xhi[ROWS * CH];
  __shared__ alignas(32) _Float16 s_xlo[ROWS * CH];
  __shared__ alignas(32) _Float16 s_whi[ROWS * 2 * CH];
  __shared__ alignas(32) _Float16 s_wlo[ROWS * 2 * CH];
  __shared__ int   s_pair[ROWS];
  __shared__ int   s_a[ROWS];
  __shared__ int   s_b[ROWS];
  __shared__ float s_dot[3 * ROWS];   // [0..15]=x.W0 [16..31]=x.W1 [32..47]=|x|^2

  const int tid  = threadIdx.x;
  const int wave = tid >> 5;
  const int lane = tid & 31;
  const int half = lane >> 4;   // K-half selector within a wave fragment
  const int sub  = lane & 15;   // row (A) / column (B) index in the tile
  const int tile = blockIdx.x * ROWS;

  // ---- Phase 1: top-2 of first_out. wave w -> rows 2w, 2w+1 (16 lanes each)
  {
    const int row  = wave * 2 + half;
    const float* f = fo + (size_t)(tile + row) * NCLS;
    float v1 = -INFINITY, v2 = -INFINITY;
    int   i1 = 0x7fffffff, i2 = 0x7fffffff;
    for (int k = sub; k < NCLS; k += 16) {
      float v = f[k];
      if (v > v1 || (v == v1 && k < i1)) { v2 = v1; i2 = i1; v1 = v; i1 = k; }
      else if (v > v2 || (v == v2 && k < i2)) { v2 = v; i2 = k; }
    }
    #pragma unroll
    for (int m = 1; m <= 8; m <<= 1) {
      float rv1 = __shfl_xor(v1, m, 16); int ri1 = __shfl_xor(i1, m, 16);
      float rv2 = __shfl_xor(v2, m, 16); int ri2 = __shfl_xor(i2, m, 16);
      if (rv1 > v1 || (rv1 == v1 && ri1 < i1)) { v2 = v1; i2 = i1; v1 = rv1; i1 = ri1; }
      else if (rv1 > v2 || (rv1 == v2 && ri1 < i2)) { v2 = rv1; i2 = ri1; }
      if (rv2 > v1 || (rv2 == v1 && ri2 < i1)) { v2 = v1; i2 = i1; v1 = rv2; i1 = ri2; }
      else if (rv2 > v2 || (rv2 == v2 && ri2 < i2)) { v2 = rv2; i2 = ri2; }
    }
    if (sub == 0) {
      int a = min(i1, i2), b = max(i1, i2);
      s_a[row] = a; s_b[row] = b;
      s_pair[row] = b * (b - 1) / 2 + a;
    }
  }
  if (tid < 3 * ROWS) s_dot[tid] = 0.f;
  __syncthreads();

  // hoist pair indices into registers (constant for the whole tile)
  int pr[ROWS];
  #pragma unroll
  for (int r = 0; r < ROWS; ++r) pr[r] = s_pair[r];

  // staging decomposition: thread -> (row-in-group, column-quad)
  const int cq = tid & 63;   // column quad: covers 4*cq .. 4*cq+3 of a 256-col chunk
  const int rq = tid >> 6;   // row selector within a 4-row group

  // ---- Phase 2: K-chunked staging + WMMA
  v8f c0 = {}, c1 = {}, c2 = {};
  for (int ch = 0; ch < NCH; ++ch) {
    const int k0 = ch * CH;

    // issue ALL staging loads first: 4 x-quads (non-temporal stream) + 8 W-quads
    v4f vx[4];
    #pragma unroll
    for (int mm = 0; mm < 4; ++mm) {
      const int m = mm * 4 + rq;
      vx[mm] = __builtin_nontemporal_load(
                   (const v4f*)&x[(size_t)(tile + m) * DIM + k0 + cq * 4]);
    }
    v4f vw[8];
    #pragma unroll
    for (int mm = 0; mm < 8; ++mm) {
      const int m = mm * 4 + rq;        // w-row 0..31
      const int r = m >> 1, j = m & 1;
      vw[mm] = *(const v4f*)&W[(size_t)pr[r] * (2 * DIM) + (size_t)j * DIM + k0 + cq * 4];
    }
    // convert + store as loads drain (descending loadcnt waits)
    #pragma unroll
    for (int mm = 0; mm < 4; ++mm) {
      const int m = mm * 4 + rq;
      uint2 hi, lo;
      split4(vx[mm], hi, lo);
      ((uint2*)s_xhi)[(m * CH) / 4 + cq] = hi;
      ((uint2*)s_xlo)[(m * CH) / 4 + cq] = lo;
    }
    #pragma unroll
    for (int mm = 0; mm < 8; ++mm) {
      const int m = mm * 4 + rq;
      uint2 hi, lo;
      split4(vw[mm], hi, lo);
      ((uint2*)s_whi)[(m * CH) / 4 + cq] = hi;
      ((uint2*)s_wlo)[(m * CH) / 4 + cq] = lo;
    }
    // prefetch next chunk while this chunk's barrier+WMMA phase runs
    if (ch + 1 < NCH) {
      __builtin_prefetch(&x[(size_t)(tile + rq) * DIM + k0 + CH + cq * 4], 0, 0);
      __builtin_prefetch(&W[(size_t)pr[rq >> 1] * (2 * DIM) + (size_t)(rq & 1) * DIM
                            + k0 + CH + cq * 4], 0, 0);
    }
    __syncthreads();

    // wave owns K-window [kb, kb+32) of this chunk
    const int kb = wave * 32;
    // A fragment (16-bit A 16x32): lane row = sub; K = kb + half*16 + 0..15 contiguous
    const int as = half * 16;
    v8h a0 = *(const v8h*)&s_xhi[sub * CH + kb + as];
    v8h a1 = *(const v8h*)&s_xhi[sub * CH + kb + 8 + as];
    v16h ahi = __builtin_shufflevector(a0, a1, 0,1,2,3,4,5,6,7,8,9,10,11,12,13,14,15);
    a0 = *(const v8h*)&s_xlo[sub * CH + kb + as];
    a1 = *(const v8h*)&s_xlo[sub * CH + kb + 8 + as];
    v16h alo = __builtin_shufflevector(a0, a1, 0,1,2,3,4,5,6,7,8,9,10,11,12,13,14,15);
    // B fragments (16-bit B 32x16): column = sub, K = kb + half*16 + 0..15 contiguous
    const int bo = kb + half * 16;
    v16h b0h = *(const v16h*)&s_whi[(sub * 2 + 0) * CH + bo];
    v16h b0l = *(const v16h*)&s_wlo[(sub * 2 + 0) * CH + bo];
    v16h b1h = *(const v16h*)&s_whi[(sub * 2 + 1) * CH + bo];
    v16h b1l = *(const v16h*)&s_wlo[(sub * 2 + 1) * CH + bo];
    v16h bxh = *(const v16h*)&s_xhi[sub * CH + bo];
    v16h bxl = *(const v16h*)&s_xlo[sub * CH + bo];

    // hi*hi + hi*lo + lo*hi  (f32-quality dot from f16 WMMA)
    c0 = wmma16(ahi, b0h, c0); c0 = wmma16(ahi, b0l, c0); c0 = wmma16(alo, b0h, c0);
    c1 = wmma16(ahi, b1h, c1); c1 = wmma16(ahi, b1l, c1); c1 = wmma16(alo, b1h, c1);
    c2 = wmma16(ahi, bxh, c2); c2 = wmma16(ahi, bxl, c2); c2 = wmma16(alo, bxh, c2);
    __syncthreads();
  }

  // ---- Phase 3: extract diagonals, reduce partials across the 8 waves
  {
    const int n    = sub;
    const int moff = half * 8;   // C/D layout: vgpr v, lanes>=16 -> M = v+8
    #pragma unroll
    for (int v = 0; v < 8; ++v) {
      if (v + moff == n) {
        atomicAdd(&s_dot[0 * ROWS + n], c0[v]);
        atomicAdd(&s_dot[1 * ROWS + n], c1[v]);
        atomicAdd(&s_dot[2 * ROWS + n], c2[v]);
      }
    }
  }
  __syncthreads();

  // ---- Phase 4: finalize (one thread per row)
  if (tid < ROWS) {
    const int r = tid;
    const int g = tile + r;
    float d0  = s_dot[r];
    float d1  = s_dot[ROWS + r];
    float xn2 = s_dot[2 * ROWS + r];
    float inv = 1.f / fmaxf(sqrtf(xn2), EPS);
    int   p   = s_pair[r];
    float l0  = scale[p * 2 + 0] * d0 * inv;
    float l1  = scale[p * 2 + 1] * d1 * inv;
    int pred  = (l1 > l0) ? s_b[r] : s_a[r];   // argmax ties -> index 0 -> a
    out[g] = (float)pred;
    out[(size_t)B + (size_t)g * 2 + 0] = l0;
    out[(size_t)B + (size_t)g * 2 + 1] = l1;
  }
}

// ---------------------------------------------------------------------------
extern "C" void kernel_launch(void* const* d_in, const int* in_sizes, int n_in,
                              void* d_out, int out_size, void* d_ws, size_t ws_size,
                              hipStream_t stream) {
  const float* x     = (const float*)d_in[0];
  const float* fo    = (const float*)d_in[1];
  const float* W     = (const float*)d_in[2];
  const float* sigma = (const float*)d_in[3];
  (void)n_in; (void)out_size; (void)ws_size;

  const int B = in_sizes[0] / DIM;        // 65536
  const int P = in_sizes[3];              // 4950
  const int nrows = P * 2;                // 9900 weight rows

  float* scale = (float*)d_ws;            // nrows floats (~40 KB) of scratch

  wscale_kernel<<<(nrows + 7) / 8, 256, 0, stream>>>(W, sigma, scale, nrows);
  mlcl_kernel<<<B / ROWS, 256, 0, stream>>>(x, fo, W, scale, (float*)d_out, B);
}
